// SLM_34016140984360
// MI455X (gfx1250) — matrix operand order repaired and checked by reading, not compiled, not run on hardware
//
#include <hip/hip_runtime.h>
#include <hip/hip_bf16.h>

// ---------------- model constants ----------------
#define B_   32
#define P_   256
#define L_   32
#define V_   30000
#define DE_  256
#define DP_  256
#define NH_  8
#define DH_  32
#define DFF_ 1024
#define NL_  4

typedef __attribute__((ext_vector_type(16))) __bf16 v16bf;
typedef __attribute__((ext_vector_type(8)))  float  v8f;

#define DEV static __device__ __forceinline__

DEV unsigned short f2bf_u16(float f) {
    unsigned u = __float_as_uint(f);
    u += 0x7FFFu + ((u >> 16) & 1u);     // round-to-nearest-even
    return (unsigned short)(u >> 16);
}
DEV float sigm(float x) { return 1.f / (1.f + __expf(-x)); }

DEV v8f vzero() {
    v8f z;
    #pragma unroll
    for (int i = 0; i < 8; i++) z[i] = 0.f;
    return z;
}

DEV v8f wmma_bf16(v16bf a, v16bf b, v8f c) {
    return __builtin_amdgcn_wmma_f32_16x16x32_bf16(
        false, a, false, b, (short)0, c, false, false);
}

// Opaque zero: defeats LICM (so loop-invariant weight-fragment loads are
// re-issued per iteration, staying L2-resident instead of hoist+spill to
// scratch) WITHOUT laundering pointer provenance -> loads stay global_load.
DEV size_t opaque_zero() {
    size_t z = 0;
    asm volatile("" : "+s"(z));
    return z;
}

// A-fragment (16x32 bf16) from row-major u16 source. base points at (m0,k0).
// Lane l<16: rows 0..15, K = {hi*16+0..15}  (contiguous layout)
DEV v16bf load_a_rm(const unsigned short* base, int stride, int lane) {
    int row = lane & 15;
    int hi  = (lane >> 4) & 1;
    const unsigned short* p = base + row * stride + hi * 16;
    union { v16bf v; uint4 q[2]; } u;
    u.q[0] = *reinterpret_cast<const uint4*>(p);
    u.q[1] = *reinterpret_cast<const uint4*>(p + 8);
    return u.v;
}

// B-fragment (32x16) from pre-packed buffer: tile = 512 contiguous halves.
DEV v16bf load_b_packed(const unsigned short* pack, size_t tile_idx, int lane) {
    const unsigned short* p = pack + (tile_idx * 32 + (size_t)lane) * 16;
    union { v16bf v; uint4 q[2]; } u;
    u.q[0] = *reinterpret_cast<const uint4*>(p);
    u.q[1] = *reinterpret_cast<const uint4*>(p + 8);
    return u.v;
}

// B-fragment (32x16) where the source is B^T row-major [N,K]; base at (n0,k0).
// b[j] = Bt[n0+col][k0 + j + hi*16]  -> fully contiguous per lane.
DEV v16bf load_b_t_rm(const unsigned short* base, int stride, int lane) {
    int col = lane & 15;
    int hi  = (lane >> 4) & 1;
    const unsigned short* p = base + col * stride + hi * 16;
    union { v16bf v; uint4 q[2]; } u;
    u.q[0] = *reinterpret_cast<const uint4*>(p);
    u.q[1] = *reinterpret_cast<const uint4*>(p + 8);
    return u.v;
}

// ---------------- weight packing ----------------
__global__ void cvt_bf_kernel(const float* __restrict__ in,
                              unsigned short* __restrict__ out, int n) {
    int i = blockIdx.x * 256 + threadIdx.x;
    if (i < n) out[i] = f2bf_u16(in[i]);
}

// Pack f32 row-major W[K][N] into bf16 WMMA B-fragment tiles.
__global__ void pack_b_kernel(const float* __restrict__ W,
                              unsigned short* __restrict__ out, int K, int N) {
    int o = blockIdx.x * 256 + threadIdx.x;
    if (o >= K * N) return;
    int j    = o & 15;
    int lane = (o >> 4) & 31;
    int tile = o >> 9;
    int ntc  = N >> 4;
    int nt   = tile % ntc;
    int kt   = tile / ntc;
    int hi   = lane >> 4, col = lane & 15;
    int rowk = kt * 32 + j + hi * 16;
    int coln = nt * 16 + col;
    out[o] = f2bf_u16(W[(size_t)rowk * N + coln]);
}

// ---------------- fused embed + LSTM1 + LSTM2 ----------------
// 16 paths per block, 8 waves. Wave w owns units [32w,32w+32) (2 unit-tiles),
// all 4 gates, as 16x16 C-layout fragments. Cell state stays in registers.
__global__ __launch_bounds__(256) void lstm_kernel(
    const int* __restrict__ tokens, const unsigned short* __restrict__ embbf,
    const unsigned short* __restrict__ K1pi, const unsigned short* __restrict__ R1pi,
    const unsigned short* __restrict__ K2pi, const unsigned short* __restrict__ R2pi,
    const float* __restrict__ b1v, const float* __restrict__ b2v,
    float* __restrict__ Hall) {
    __shared__ unsigned short xbf[16 * 256];
    __shared__ unsigned short h1bf[16 * 256];
    __shared__ unsigned short h2bf[16 * 256];

    const int tid  = threadIdx.x;
    const int lane = tid & 31;
    const int w    = tid >> 5;
    const int col  = lane & 15;
    const int hi   = (lane >> 4) & 1;
    const int blockPath = blockIdx.x * 16;

    for (int i = tid; i < 16 * 256; i += 256) { h1bf[i] = 0; h2bf[i] = 0; }

    v8f c1[2], c2[2];
    for (int u = 0; u < 2; u++) { c1[u] = vzero(); c2[u] = vzero(); }

    float bia1[2][4], bia2[2][4];
    for (int u = 0; u < 2; u++)
        for (int g = 0; g < 4; g++) {
            int unit = w * 32 + u * 16 + col;
            bia1[u][g] = b1v[g * 256 + unit];
            bia2[u][g] = b2v[g * 256 + unit];
        }
    __syncthreads();

    for (int t = 0; t < L_; t++) {
        // Per-iteration opaque zero offset: fragment loads cannot be hoisted
        // out of the t-loop (no spills), but pointers keep global provenance.
        size_t zz = opaque_zero();
        const unsigned short* K1p = K1pi + zz;
        const unsigned short* R1p = R1pi + zz;
        const unsigned short* K2p = K2pi + zz;
        const unsigned short* R2p = R2pi + zz;

        // gather x_t rows from embedding (bf16), 32B per thread chunk
        {
            int p   = tid >> 4;
            int seg = tid & 15;
            int tok = tokens[(blockPath + p) * L_ + t];
            const unsigned short* src = embbf + (size_t)tok * 256 + seg * 16;
            uint4 d0 = *reinterpret_cast<const uint4*>(src);
            uint4 d1 = *reinterpret_cast<const uint4*>(src + 8);
            *reinterpret_cast<uint4*>(&xbf[p * 256 + seg * 16])     = d0;
            *reinterpret_cast<uint4*>(&xbf[p * 256 + seg * 16 + 8]) = d1;
            if (t + 1 < L_) {
                int tok2 = tokens[(blockPath + p) * L_ + t + 1];
                __builtin_prefetch(embbf + (size_t)tok2 * 256, 0, 0);
            }
        }
        __syncthreads();

        // ---- LSTM1: z = x@K1 + h1@R1 ----
        v8f acc[2][4];
        for (int u = 0; u < 2; u++) for (int g = 0; g < 4; g++) acc[u][g] = vzero();
        #pragma unroll
        for (int kc = 0; kc < 8; kc++) {
            v16bf a = load_a_rm(xbf + kc * 32, 256, lane);
            for (int u = 0; u < 2; u++)
                for (int g = 0; g < 4; g++) {
                    int nt = g * 16 + w * 2 + u;
                    v16bf bb = load_b_packed(K1p, (size_t)kc * 64 + nt, lane);
                    acc[u][g] = wmma_bf16(a, bb, acc[u][g]);
                }
        }
        #pragma unroll
        for (int kc = 0; kc < 8; kc++) {
            v16bf a = load_a_rm(h1bf + kc * 32, 256, lane);
            for (int u = 0; u < 2; u++)
                for (int g = 0; g < 4; g++) {
                    int nt = g * 16 + w * 2 + u;
                    v16bf bb = load_b_packed(R1p, (size_t)kc * 64 + nt, lane);
                    acc[u][g] = wmma_bf16(a, bb, acc[u][g]);
                }
        }
        __syncthreads();   // all waves done reading h1bf(t-1)

        for (int u = 0; u < 2; u++) {
            int unit = w * 32 + u * 16 + col;
            #pragma unroll
            for (int r = 0; r < 8; r++) {
                float iv = acc[u][0][r] + bia1[u][0];
                float fv = acc[u][1][r] + bia1[u][1];
                float gv = acc[u][2][r] + bia1[u][2];
                float ov = acc[u][3][r] + bia1[u][3];
                float c  = sigm(fv) * c1[u][r] + sigm(iv) * tanhf(gv);
                float h  = sigm(ov) * tanhf(c);
                c1[u][r] = c;
                int row = r + hi * 8;
                h1bf[row * 256 + unit] = f2bf_u16(h);
            }
        }
        __syncthreads();

        // ---- LSTM2: z = h1@K2 + h2@R2 ----
        for (int u = 0; u < 2; u++) for (int g = 0; g < 4; g++) acc[u][g] = vzero();
        #pragma unroll
        for (int kc = 0; kc < 8; kc++) {
            v16bf a = load_a_rm(h1bf + kc * 32, 256, lane);
            for (int u = 0; u < 2; u++)
                for (int g = 0; g < 4; g++) {
                    int nt = g * 16 + w * 2 + u;
                    v16bf bb = load_b_packed(K2p, (size_t)kc * 64 + nt, lane);
                    acc[u][g] = wmma_bf16(a, bb, acc[u][g]);
                }
        }
        #pragma unroll
        for (int kc = 0; kc < 8; kc++) {
            v16bf a = load_a_rm(h2bf + kc * 32, 256, lane);
            for (int u = 0; u < 2; u++)
                for (int g = 0; g < 4; g++) {
                    int nt = g * 16 + w * 2 + u;
                    v16bf bb = load_b_packed(R2p, (size_t)kc * 64 + nt, lane);
                    acc[u][g] = wmma_bf16(a, bb, acc[u][g]);
                }
        }
        __syncthreads();

        for (int u = 0; u < 2; u++) {
            int unit = w * 32 + u * 16 + col;
            #pragma unroll
            for (int r = 0; r < 8; r++) {
                float iv = acc[u][0][r] + bia2[u][0];
                float fv = acc[u][1][r] + bia2[u][1];
                float gv = acc[u][2][r] + bia2[u][2];
                float ov = acc[u][3][r] + bia2[u][3];
                float c  = sigm(fv) * c2[u][r] + sigm(iv) * tanhf(gv);
                float h  = sigm(ov) * tanhf(c);
                c2[u][r] = c;
                int row = r + hi * 8;
                h2bf[row * 256 + unit] = f2bf_u16(h);
                if (t == L_ - 1)
                    Hall[(size_t)(blockPath + row) * 256 + unit] = h;
            }
        }
        __syncthreads();
    }
}

// ---------------- encoder input build ----------------
__global__ void xbuild_kernel(const float* __restrict__ Hall,
                              const int* __restrict__ pc,
                              float* __restrict__ Xf,
                              unsigned short* __restrict__ Xb) {
    int idx = blockIdx.x * 256 + threadIdx.x;     // over 8192*256
    int row = idx >> 8;
    int bs  = row >> 8;
    int i   = row & 255;
    int vlen = pc[bs] - 1;
    float v = (i < vlen) ? Hall[idx] : 0.f;
    Xf[idx] = v;
    Xb[idx] = f2bf_u16(v);
}

// ---------------- generic packed-B WMMA GEMM ----------------
__global__ __launch_bounds__(256) void gemm_pk_kernel(
    const unsigned short* __restrict__ A, const unsigned short* __restrict__ Bp,
    const float* __restrict__ bias, float* __restrict__ Cf,
    unsigned short* __restrict__ Cb, int M, int N, int K, int relu) {
    int tid = threadIdx.x, lane = tid & 31, w = tid >> 5;
    int wid = blockIdx.x * 8 + w;
    int Nt = N >> 4, Kt = K >> 5;
    if (wid >= (M >> 4) * Nt) return;
    int mt = wid / Nt, nt = wid % Nt;
    int col = lane & 15, hi = (lane >> 4) & 1;
    v8f acc = vzero();
    const unsigned short* Abase = A + (size_t)mt * 16 * K;
    for (int kt = 0; kt < Kt; kt++) {
        v16bf a  = load_a_rm(Abase + kt * 32, K, lane);
        v16bf bb = load_b_packed(Bp, (size_t)kt * Nt + nt, lane);
        acc = wmma_bf16(a, bb, acc);
    }
    int n = nt * 16 + col;
    float bv = bias ? bias[n] : 0.f;
    #pragma unroll
    for (int r = 0; r < 8; r++) {
        int m = mt * 16 + r + hi * 8;
        float v = acc[r] + bv;
        if (relu) v = fmaxf(v, 0.f);
        if (Cf) Cf[(size_t)m * N + n] = v;
        if (Cb) Cb[(size_t)m * N + n] = f2bf_u16(v);
    }
}

// GEMM with B given as B^T row-major [N,K] (classifier: logits = h @ emb^T).
__global__ __launch_bounds__(256) void gemm_bt_kernel(
    const unsigned short* __restrict__ A, const unsigned short* __restrict__ Bt,
    float* __restrict__ Cf, int M, int N, int K) {
    int tid = threadIdx.x, lane = tid & 31, w = tid >> 5;
    int wid = blockIdx.x * 8 + w;
    int Nt = N >> 4, Kt = K >> 5;
    if (wid >= (M >> 4) * Nt) return;
    int mt = wid / Nt, nt = wid % Nt;
    int col = lane & 15, hi = (lane >> 4) & 1;
    v8f acc = vzero();
    const unsigned short* Abase = A + (size_t)mt * 16 * K;
    for (int kt = 0; kt < Kt; kt++) {
        v16bf a  = load_a_rm(Abase + kt * 32, K, lane);
        v16bf bb = load_b_t_rm(Bt + (size_t)nt * 16 * K + kt * 32, K, lane);
        acc = wmma_bf16(a, bb, acc);
    }
    int n = nt * 16 + col;
    #pragma unroll
    for (int r = 0; r < 8; r++) {
        int m = mt * 16 + r + hi * 8;
        Cf[(size_t)m * N + n] = acc[r];
    }
}

// ---------------- flash attention (one wave per (b, head, q-tile)) ----------------
__global__ __launch_bounds__(256) void attn_kernel(
    const unsigned short* __restrict__ Qbf, const unsigned short* __restrict__ Kbf,
    const unsigned short* __restrict__ Vbf, const int* __restrict__ pc,
    unsigned short* __restrict__ Obf) {
    __shared__ unsigned short pstage[8 * 16 * 32];
    int tid = threadIdx.x, lane = tid & 31, w = tid >> 5;
    int wid = blockIdx.x * 8 + w;           // 0..4095
    int b  = wid >> 7;
    int h  = (wid >> 4) & 7;
    int qt = wid & 15;
    int vlen = pc[b] - 1;
    const int base = b * 256;
    int col = lane & 15, hi = (lane >> 4) & 1;
    const float scale = 0.17677669529663687f;   // 1/sqrt(DH)

    v16bf qf = load_a_rm(Qbf + (size_t)(base + qt * 16) * 256 + h * 32, 256, lane);
    v8f o0 = vzero(), o1 = vzero();
    float m8[8], l8[8];
    #pragma unroll
    for (int r = 0; r < 8; r++) { m8[r] = -3.0e38f; l8[r] = 0.f; }

    unsigned short* ps = pstage + w * 512;

    for (int kc = 0; kc < 8; kc++) {
        int k0 = kc * 32;
        v16bf kf0 = load_b_t_rm(Kbf + (size_t)(base + k0) * 256 + h * 32, 256, lane);
        v16bf kf1 = load_b_t_rm(Kbf + (size_t)(base + k0 + 16) * 256 + h * 32, 256, lane);
        v8f s0 = wmma_bf16(qf, kf0, vzero());
        v8f s1 = wmma_bf16(qf, kf1, vzero());
        bool kv0 = (k0 + col)      < vlen;
        bool kv1 = (k0 + 16 + col) < vlen;
        #pragma unroll
        for (int r = 0; r < 8; r++) {
            int q = qt * 16 + r + hi * 8;
            bool qv = q < vlen;
            float sv0 = (qv && kv0) ? s0[r] * scale : -1e9f;
            float sv1 = (qv && kv1) ? s1[r] * scale : -1e9f;
            float mx = fmaxf(sv0, sv1);
            for (int d = 8; d >= 1; d >>= 1) mx = fmaxf(mx, __shfl_xor(mx, d, 32));
            float mnew = fmaxf(m8[r], mx);
            float fs = __expf(m8[r] - mnew);
            float p0 = __expf(sv0 - mnew);
            float p1 = __expf(sv1 - mnew);
            float ls = p0 + p1;
            for (int d = 8; d >= 1; d >>= 1) ls += __shfl_xor(ls, d, 32);
            l8[r] = l8[r] * fs + ls;
            o0[r] *= fs; o1[r] *= fs;
            m8[r] = mnew;
            int row = r + hi * 8;
            ps[row * 32 + col]      = f2bf_u16(p0);
            ps[row * 32 + 16 + col] = f2bf_u16(p1);
        }
        asm volatile("s_wait_dscnt 0" ::: "memory");   // P tile staged in LDS
        v16bf pf = load_a_rm(ps, 32, lane);
        v16bf vf0, vf1;
        {
            union { v16bf v; unsigned short hh[16]; } u0, u1;
            #pragma unroll
            for (int j = 0; j < 16; j++) {
                int krow = k0 + j + hi * 16;
                const unsigned short* vp = Vbf + (size_t)(base + krow) * 256 + h * 32;
                u0.hh[j] = vp[col];
                u1.hh[j] = vp[16 + col];
            }
            vf0 = u0.v; vf1 = u1.v;
        }
        o0 = wmma_bf16(pf, vf0, o0);
        o1 = wmma_bf16(pf, vf1, o1);
    }
    #pragma unroll
    for (int r = 0; r < 8; r++) {
        float inv = 1.f / l8[r];
        int row = qt * 16 + r + hi * 8;
        unsigned short* op = Obf + (size_t)(base + row) * 256 + h * 32;
        op[col]      = f2bf_u16(o0[r] * inv);
        op[16 + col] = f2bf_u16(o1[r] * inv);
    }
}

// ---------------- residual add + LayerNorm (wave per row) ----------------
__global__ __launch_bounds__(256) void add_ln_kernel(
    const float* __restrict__ x, const float* __restrict__ y,
    const float* __restrict__ g, const float* __restrict__ bb,
    float* __restrict__ xo, unsigned short* __restrict__ xb, int rows) {
    int tid = threadIdx.x, lane = tid & 31, w = tid >> 5;
    int row = blockIdx.x * 8 + w;
    if (row >= rows) return;
    const float* xr = x + (size_t)row * 256;
    const float* yr = y + (size_t)row * 256;
    float v[8];
    float s = 0.f;
    #pragma unroll
    for (int i = 0; i < 8; i++) { v[i] = xr[lane + i * 32] + yr[lane + i * 32]; s += v[i]; }
    for (int d = 16; d >= 1; d >>= 1) s += __shfl_xor(s, d, 32);
    float mean = s * (1.f / 256.f);
    float vs = 0.f;
    #pragma unroll
    for (int i = 0; i < 8; i++) { float t = v[i] - mean; vs += t * t; }
    for (int d = 16; d >= 1; d >>= 1) vs += __shfl_xor(vs, d, 32);
    float rstd = rsqrtf(vs * (1.f / 256.f) + 1e-5f);
    #pragma unroll
    for (int i = 0; i < 8; i++) {
        int c = lane + i * 32;
        float o = (v[i] - mean) * rstd * g[c] + bb[c];
        xo[(size_t)row * 256 + c] = o;
        xb[(size_t)row * 256 + c] = f2bf_u16(o);
    }
}

// ---------------- pooling ----------------
__global__ __launch_bounds__(256) void pool_r_kernel(
    const float* __restrict__ Hall, const int* __restrict__ pc,
    const int* __restrict__ ti, const float* __restrict__ Ci,
    const float* __restrict__ Wr, float* __restrict__ Rout) {
    __shared__ float R0[256], Rt[256];
    int b = blockIdx.x, t = threadIdx.x;
    int p = pc[b] - 1;
    R0[t] = Hall[(size_t)(b * 256 + p) * 256 + t];
    __syncthreads();
    const float* C = Ci + (size_t)ti[b] * 256 * 256;
    float s = 0.f;
    for (int j = 0; j < 256; j++) s += C[t * 256 + j] * R0[j];
    Rt[t] = fmaxf(s, 0.f);
    __syncthreads();
    float s2 = 0.f;
    for (int j = 0; j < 256; j++) s2 += Rt[j] * Wr[(size_t)j * 256 + t];
    Rout[b * 256 + t] = s2;
}

__global__ __launch_bounds__(256) void pool_alpha_kernel(
    const float* __restrict__ Z, const float* __restrict__ Rv,
    const float* __restrict__ Wg, unsigned short* __restrict__ hbf) {
    __shared__ float R[256], sc[256], Zp[256], red[256];
    int b = blockIdx.x, t = threadIdx.x;
    R[t] = Rv[b * 256 + t];
    __syncthreads();
    float s = -1e30f;
    if (t < 255) {
        const float* zr = Z + (size_t)(b * 256 + t) * 256;
        float d = 0.f;
        for (int j = 0; j < 256; j++) d += zr[j] * R[j];
        s = d;
    }
    sc[t] = s; red[t] = s;
    __syncthreads();
    for (int st = 128; st >= 1; st >>= 1) { if (t < st) red[t] = fmaxf(red[t], red[t + st]); __syncthreads(); }
    float mx = red[0];
    __syncthreads();
    float e = (t < 255) ? __expf(sc[t] - mx) : 0.f;
    sc[t] = e; red[t] = e;
    __syncthreads();
    for (int st = 128; st >= 1; st >>= 1) { if (t < st) red[t] += red[t + st]; __syncthreads(); }
    float inv = 1.f / red[0];
    __syncthreads();
    float acc = 0.f;
    for (int n = 0; n < 255; n++) acc += sc[n] * Z[(size_t)(b * 256 + n) * 256 + t];
    Zp[t] = acc * inv;
    __syncthreads();
    float hh = 0.f;
    for (int j = 0; j < 256; j++) hh += Zp[j] * Wg[(size_t)j * 256 + t];
    for (int j = 0; j < 256; j++) hh += R[j] * Wg[(size_t)(256 + j) * 256 + t];
    hbf[b * 256 + t] = f2bf_u16(fmaxf(hh, 0.f));
}

__global__ __launch_bounds__(256) void vsoftmax_kernel(
    const float* __restrict__ logits, float* __restrict__ out) {
    __shared__ float red[256];
    int b = blockIdx.x, t = threadIdx.x;
    const float* lr = logits + (size_t)b * V_;
    float mx = -3e38f;
    for (int v = t; v < V_; v += 256) mx = fmaxf(mx, lr[v]);
    red[t] = mx; __syncthreads();
    for (int st = 128; st >= 1; st >>= 1) { if (t < st) red[t] = fmaxf(red[t], red[t + st]); __syncthreads(); }
    mx = red[0]; __syncthreads();
    float s = 0.f;
    for (int v = t; v < V_; v += 256) s += __expf(lr[v] - mx);
    red[t] = s; __syncthreads();
    for (int st = 128; st >= 1; st >>= 1) { if (t < st) red[t] += red[t + st]; __syncthreads(); }
    float inv = 1.f / red[0];
    float* orow = out + (size_t)b * V_;
    for (int v = t; v < V_; v += 256) orow[v] = __expf(lr[v] - mx) * inv;
}

// ---------------- host orchestration ----------------
extern "C" void kernel_launch(void* const* d_in, const int* in_sizes, int n_in,
                              void* d_out, int out_size, void* d_ws, size_t ws_size,
                              hipStream_t stream) {
    (void)in_sizes; (void)n_in; (void)out_size; (void)ws_size;
    const int*   tokens = (const int*)d_in[0];
    const int*   pc     = (const int*)d_in[1];
    const int*   ti     = (const int*)d_in[2];
    const float* emb    = (const float*)d_in[3];
    const float* l1k = (const float*)d_in[4];
    const float* l1r = (const float*)d_in[5];
    const float* l1b = (const float*)d_in[6];
    const float* l2k = (const float*)d_in[7];
    const float* l2r = (const float*)d_in[8];
    const float* l2b = (const float*)d_in[9];
    const float* wq  = (const float*)d_in[10];
    const float* wk  = (const float*)d_in[11];
    const float* wv  = (const float*)d_in[12];
    const float* wo  = (const float*)d_in[13];
    const float* ln1g = (const float*)d_in[14];
    const float* ln1b = (const float*)d_in[15];
    const float* w1  = (const float*)d_in[16];
    const float* b1  = (const float*)d_in[17];
    const float* w2  = (const float*)d_in[18];
    const float* b2  = (const float*)d_in[19];
    const float* ln2g = (const float*)d_in[20];
    const float* ln2b = (const float*)d_in[21];
    const float* Wr  = (const float*)d_in[22];
    const float* Wg  = (const float*)d_in[23];
    const float* Ci  = (const float*)d_in[24];
    float* out = (float*)d_out;

    char* ws = (char*)d_ws;
    size_t off = 0;
    auto alloc = [&](size_t bytes) -> char* {
        char* p = ws + off;
        off = (off + bytes + 255) & ~(size_t)255;
        return p;
    };
    const int ROWS = B_ * P_;   // 8192

    unsigned short* embbf = (unsigned short*)alloc((size_t)V_ * DE_ * 2);
    unsigned short* K1p = (unsigned short*)alloc(256 * 1024 * 2);
    unsigned short* R1p = (unsigned short*)alloc(256 * 1024 * 2);
    unsigned short* K2p = (unsigned short*)alloc(256 * 1024 * 2);
    unsigned short* R2p = (unsigned short*)alloc(256 * 1024 * 2);
    unsigned short* wqp = (unsigned short*)alloc((size_t)NL_ * 65536 * 2);
    unsigned short* wkp = (unsigned short*)alloc((size_t)NL_ * 65536 * 2);
    unsigned short* wvp = (unsigned short*)alloc((size_t)NL_ * 65536 * 2);
    unsigned short* wop = (unsigned short*)alloc((size_t)NL_ * 65536 * 2);
    unsigned short* w1p = (unsigned short*)alloc((size_t)NL_ * 262144 * 2);
    unsigned short* w2p = (unsigned short*)alloc((size_t)NL_ * 262144 * 2);
    float*          Hall = (float*)alloc((size_t)ROWS * 256 * 4);
    float*          Xf   = (float*)alloc((size_t)ROWS * 256 * 4);
    unsigned short* Xb   = (unsigned short*)alloc((size_t)ROWS * 256 * 2);
    unsigned short* Qb   = (unsigned short*)alloc((size_t)ROWS * 256 * 2);
    unsigned short* Kb   = (unsigned short*)alloc((size_t)ROWS * 256 * 2);
    unsigned short* Vb   = (unsigned short*)alloc((size_t)ROWS * 256 * 2);
    unsigned short* Ob   = (unsigned short*)alloc((size_t)ROWS * 256 * 2);
    float*          aprj = (float*)alloc((size_t)ROWS * 256 * 4);
    unsigned short* Fb   = (unsigned short*)alloc((size_t)ROWS * 1024 * 2);
    float*          ffo  = (float*)alloc((size_t)ROWS * 256 * 4);
    float*          Rv   = (float*)alloc((size_t)B_ * 256 * 4);
    unsigned short* hbf  = (unsigned short*)alloc((size_t)B_ * 256 * 2);
    float*          logits = (float*)alloc((size_t)B_ * V_ * 4);

    // ---- weight packing / conversion ----
    cvt_bf_kernel<<<(V_ * DE_ + 255) / 256, 256, 0, stream>>>(emb, embbf, V_ * DE_);
    pack_b_kernel<<<1024, 256, 0, stream>>>(l1k, K1p, 256, 1024);
    pack_b_kernel<<<1024, 256, 0, stream>>>(l1r, R1p, 256, 1024);
    pack_b_kernel<<<1024, 256, 0, stream>>>(l2k, K2p, 256, 1024);
    pack_b_kernel<<<1024, 256, 0, stream>>>(l2r, R2p, 256, 1024);
    for (int l = 0; l < NL_; l++) {
        pack_b_kernel<<<256, 256, 0, stream>>>(wq + (size_t)l * 65536, wqp + (size_t)l * 65536, 256, 256);
        pack_b_kernel<<<256, 256, 0, stream>>>(wk + (size_t)l * 65536, wkp + (size_t)l * 65536, 256, 256);
        pack_b_kernel<<<256, 256, 0, stream>>>(wv + (size_t)l * 65536, wvp + (size_t)l * 65536, 256, 256);
        pack_b_kernel<<<256, 256, 0, stream>>>(wo + (size_t)l * 65536, wop + (size_t)l * 65536, 256, 256);
        pack_b_kernel<<<1024, 256, 0, stream>>>(w1 + (size_t)l * 262144, w1p + (size_t)l * 262144, 256, 1024);
        pack_b_kernel<<<1024, 256, 0, stream>>>(w2 + (size_t)l * 262144, w2p + (size_t)l * 262144, 1024, 256);
    }

    // ---- fused embed + 2xLSTM ----
    lstm_kernel<<<ROWS / 16, 256, 0, stream>>>(tokens, embbf, K1p, R1p, K2p, R2p,
                                               l1b, l2b, Hall);

    // ---- encoder input (zero invalid rows) ----
    xbuild_kernel<<<ROWS, 256, 0, stream>>>(Hall, pc, Xf, Xb);

    // ---- 4 encoder layers ----
    for (int l = 0; l < NL_; l++) {
        int g256 = (ROWS / 16) * (256 / 16) / 8;    // 1024 blocks
        gemm_pk_kernel<<<g256, 256, 0, stream>>>(Xb, wqp + (size_t)l * 65536,
                                                 nullptr, nullptr, Qb, ROWS, 256, 256, 0);
        gemm_pk_kernel<<<g256, 256, 0, stream>>>(Xb, wkp + (size_t)l * 65536,
                                                 nullptr, nullptr, Kb, ROWS, 256, 256, 0);
        gemm_pk_kernel<<<g256, 256, 0, stream>>>(Xb, wvp + (size_t)l * 65536,
                                                 nullptr, nullptr, Vb, ROWS, 256, 256, 0);
        attn_kernel<<<512, 256, 0, stream>>>(Qb, Kb, Vb, pc, Ob);
        gemm_pk_kernel<<<g256, 256, 0, stream>>>(Ob, wop + (size_t)l * 65536,
                                                 nullptr, aprj, nullptr, ROWS, 256, 256, 0);
        add_ln_kernel<<<ROWS / 8, 256, 0, stream>>>(Xf, aprj, ln1g + l * 256, ln1b + l * 256,
                                                    Xf, Xb, ROWS);
        int g1024 = (ROWS / 16) * (1024 / 16) / 8;  // 4096 blocks
        gemm_pk_kernel<<<g1024, 256, 0, stream>>>(Xb, w1p + (size_t)l * 262144,
                                                  b1 + l * 1024, nullptr, Fb, ROWS, 1024, 256, 1);
        gemm_pk_kernel<<<g256, 256, 0, stream>>>(Fb, w2p + (size_t)l * 262144,
                                                 b2 + l * 256, ffo, nullptr, ROWS, 256, 1024, 0);
        add_ln_kernel<<<ROWS / 8, 256, 0, stream>>>(Xf, ffo, ln2g + l * 256, ln2b + l * 256,
                                                    Xf, Xb, ROWS);
    }

    // ---- pooling + classifier + softmax ----
    pool_r_kernel<<<B_, 256, 0, stream>>>(Hall, pc, ti, Ci, Wr, Rv);
    pool_alpha_kernel<<<B_, 256, 0, stream>>>(Xf, Rv, Wg, hbf);
    {
        int tasks = (B_ / 16) * (V_ / 16);          // 2 * 1875 = 3750 waves
        gemm_bt_kernel<<<(tasks + 7) / 8, 256, 0, stream>>>(hbf, embbf, logits, B_, V_, 256);
    }
    vsoftmax_kernel<<<B_, 256, 0, stream>>>(logits, out);
}
